// HTIHT_Cuda_10522669875892
// MI455X (gfx1250) — matrix operand order repaired and checked by reading, not compiled, not run on hardware
//
#include <hip/hip_runtime.h>

#define A_BINS 180
#define R_BINS 182
#define RP     192           // R padded to 12 tiles of 16
#define HW     16384
#define CI     64
#define CO     64
#define NB     4

typedef __attribute__((ext_vector_type(16))) __bf16 v16bf;
typedef __attribute__((ext_vector_type(8)))  __bf16 v8bf;
typedef __attribute__((ext_vector_type(8)))  float  v8f;

// ---------------- DHT: scatter x[N,C,H,W] -> ht0[n,a,r,c] (f32, channel-last) ----------------
__global__ void dht_kernel(const float* __restrict__ x,
                           const int* __restrict__ rho,
                           float* __restrict__ ht0) {
  __shared__ float acc[R_BINS * 65];          // pad 65 -> spread ds_add banks
  const int a = blockIdx.x, n = blockIdx.y, tid = threadIdx.x;
  for (int i = tid; i < R_BINS * 65; i += 256) acc[i] = 0.f;
  __syncthreads();
  const int* rp = rho + a * HW;
  const float* xn = x + (size_t)n * CI * HW;
  for (int p = tid; p < HW; p += 256) {
    const int idx = rp[p];                    // coalesced
    float* dst = acc + idx * 65;
#pragma unroll 8
    for (int c = 0; c < CI; ++c)
      atomicAdd(dst + c, xn[(size_t)c * HW + p]);   // ds_add_f32
  }
  __syncthreads();
  float* out = ht0 + (size_t)(n * A_BINS + a) * RP * 64;
  for (int i = tid; i < R_BINS * 64; i += 256) {
    const int r = i >> 6, c = i & 63;
    out[r * 64 + c] = acc[r * 65 + c];
  }
}

// ---------------- conv1: grouped (9,1) + ReLU, f32 in -> bf16 out, channel-last ----------------
__global__ void conv1_kernel(const float* __restrict__ ht0,
                             const float* __restrict__ w1,
                             const float* __restrict__ b1,
                             __bf16* __restrict__ ht1) {
  __shared__ float w1s[CI * 9];
  __shared__ float b1s[CI];
  const int a = blockIdx.x, n = blockIdx.y, tid = threadIdx.x;
  for (int i = tid; i < CI * 9; i += 256) w1s[i] = w1[i];
  if (tid < CI) b1s[tid] = b1[tid];
  __syncthreads();
  __bf16* out = ht1 + (size_t)(n * A_BINS + a) * RP * 64;
  for (int i = tid; i < RP * 64; i += 256) {
    const int r = i >> 6, c = i & 63;
    float v = 0.f;
    if (r < R_BINS) {
      float s = b1s[c];
#pragma unroll
      for (int k = 0; k < 9; ++k) {
        const int ain = a + k - 4;
        if (ain >= 0 && ain < A_BINS)
          s += w1s[c * 9 + k] * ht0[((size_t)(n * A_BINS + ain) * RP + r) * 64 + c];
      }
      v = fmaxf(s, 0.f);
    }
    out[i] = (__bf16)v;                       // pad rows (r>=182) written as 0
  }
}

// ---------------- weight prep: pre-swizzle w[CO,CI,9] into WMMA A-fragment layout ----------------
// wf[((k*2+h)*4+mt)*512 + L*16 + j] = bf16( w[co=16*mt+(L&15), ci=32*h+K(L,j), k] )
// 16-bit A 16x32 layout (contiguous k order, matches B): K(L,j) = 16*(L>>4) + j
__global__ void wprep_kernel(const float* __restrict__ w, __bf16* __restrict__ wf) {
  const int i = blockIdx.x * 256 + threadIdx.x;
  if (i >= 9 * 2 * 4 * 32 * 16) return;
  const int j  = i & 15;
  const int L  = (i >> 4) & 31;
  const int mt = (i >> 9) & 3;
  const int h  = (i >> 11) & 1;
  const int k  = i >> 12;
  const int K  = 16 * (L >> 4) + j;
  const int co = mt * 16 + (L & 15);
  const int ci = h * 32 + K;
  wf[i] = (__bf16)w[(co * CI + ci) * 9 + k];
}

// ---------------- dense (9,1) conv 64->64 + ReLU via v_wmma_f32_16x16x32_bf16 ----------------
// Block = (n,a) slab, 8 waves: wave -> (co tile mt, r-tile parity ng), 6 r-tiles each.
// B-fragment: lane L = column r0+(L&15); 16 contiguous ci bf16 = one 32B load (channel-last!).
// sched_group_barrier forces the scheduler to keep the 14-load clause (A frag + 6 B frags)
// ahead of the 6-WMMA burst, so B frags live in distinct registers and waits are partial
// (loads for later tiles overlap earlier WMMAs instead of s_wait_loadcnt 0 per WMMA).
__global__ void __launch_bounds__(256)
convd_kernel(const __bf16* __restrict__ X,
             const __bf16* __restrict__ wf,
             const float* __restrict__ bias,
             __bf16* __restrict__ Y) {
  const int a = blockIdx.x, n = blockIdx.y, tid = threadIdx.x;
  const int wv = tid >> 5, L = tid & 31;
  const int mt = wv & 3, ng = wv >> 2;
  const int lane_lo = L & 15, lane_hi = L >> 4;
  v8f acc[6] = {};
  for (int k = 0; k < 9; ++k) {
    const int ain = a + k - 4;
    if (ain < 0 || ain >= A_BINS) continue;      // uniform per block: EXEC stays all-ones
    const __bf16* xs = X + (size_t)(n * A_BINS + ain) * RP * 64
                         + (size_t)lane_lo * 64 + lane_hi * 16;
#pragma unroll
    for (int h = 0; h < 2; ++h) {
      const v16bf afr = *(const v16bf*)(wf + ((((k * 2 + h) * 4 + mt) * 32 + L) << 4));
      const __bf16* xh = xs + h * 32;
      v16bf bf[6];
#pragma unroll
      for (int t = 0; t < 6; ++t)
        bf[t] = *(const v16bf*)(xh + (size_t)(ng + 2 * t) * (16 * 64));
#pragma unroll
      for (int t = 0; t < 6; ++t)
        acc[t] = __builtin_amdgcn_wmma_f32_16x16x32_bf16(
            false, afr, false, bf[t], (short)0, acc[t], false, false);
#if __has_builtin(__builtin_amdgcn_sched_group_barrier)
      __builtin_amdgcn_sched_group_barrier(0x020, 14, 0);  // 14 VMEM reads first
      __builtin_amdgcn_sched_group_barrier(0x008, 6, 0);   // then 6 WMMAs
#endif
    }
  }
  // D layout: lane L, vgpr v -> co = 16*mt + 8*lane_hi + v (contiguous!), r = r0 + lane_lo
  const int co_base = mt * 16 + lane_hi * 8;
  float bv[8];
#pragma unroll
  for (int v = 0; v < 8; ++v) bv[v] = bias[co_base + v];
  __bf16* yb = Y + (size_t)(n * A_BINS + a) * RP * 64;
#pragma unroll
  for (int t = 0; t < 6; ++t) {
    const int r = (ng + 2 * t) * 16 + lane_lo;
    v8bf o;
#pragma unroll
    for (int v = 0; v < 8; ++v) o[v] = (__bf16)fmaxf(acc[t][v] + bv[v], 0.f);
    *(v8bf*)(yb + (size_t)r * 64 + co_base) = o;   // one 16B store per lane
  }
}

// ---------------- IDHT: gather ht3[n,a,idx,c] -> out[N,C,H,W] ----------------
__global__ void idht_kernel(const __bf16* __restrict__ ht3,
                            const int* __restrict__ rho,
                            float* __restrict__ out) {
  const int hw = blockIdx.x * 256 + threadIdx.x;
  const int c = blockIdx.y, n = blockIdx.z;
  float s = 0.f;
  for (int a = 0; a < A_BINS; ++a) {
    const int idx = rho[a * HW + hw];
    s += (float)ht3[((size_t)(n * A_BINS + a) * RP + idx) * 64 + c];
  }
  out[(size_t)(n * CO + c) * HW + hw] = s;
}

extern "C" void kernel_launch(void* const* d_in, const int* in_sizes, int n_in,
                              void* d_out, int out_size, void* d_ws, size_t ws_size,
                              hipStream_t stream) {
  (void)in_sizes; (void)n_in; (void)out_size; (void)ws_size;
  const float* x   = (const float*)d_in[0];
  const int*   rho = (const int*)d_in[1];
  const float* w1  = (const float*)d_in[2];
  const float* b1  = (const float*)d_in[3];
  const float* w2  = (const float*)d_in[4];
  const float* b2  = (const float*)d_in[5];
  const float* w3  = (const float*)d_in[6];
  const float* b3  = (const float*)d_in[7];
  float* out = (float*)d_out;

  char* ws = (char*)d_ws;
  const size_t slab = (size_t)NB * A_BINS * RP * 64;   // elements per Hough tensor
  float*  ht0 = (float*)ws;                            // slab*4 bytes
  __bf16* ht1 = (__bf16*)(ws + slab * 4);              // slab*2 bytes
  __bf16* wf2 = (__bf16*)(ws + slab * 4 + slab * 2);   // 36864 halfs
  __bf16* wf3 = wf2 + 9 * 2 * 4 * 32 * 16;
  __bf16* ht2 = (__bf16*)ws;                           // reuse ht0 slab (dead after conv1)
  __bf16* ht3 = ht2 + slab;                            // second half of ht0 slab

  wprep_kernel<<<144, 256, 0, stream>>>(w2, wf2);
  wprep_kernel<<<144, 256, 0, stream>>>(w3, wf3);
  dht_kernel  <<<dim3(A_BINS, NB), 256, 0, stream>>>(x, rho, ht0);
  conv1_kernel<<<dim3(A_BINS, NB), 256, 0, stream>>>(ht0, w1, b1, ht1);
  convd_kernel<<<dim3(A_BINS, NB), 256, 0, stream>>>(ht1, wf2, b2, ht2);
  convd_kernel<<<dim3(A_BINS, NB), 256, 0, stream>>>(ht2, wf3, b3, ht3);
  idht_kernel <<<dim3(HW / 256, CO, NB), 256, 0, stream>>>(ht3, rho, out);
}
